// ScaledDotProductAttention_46720654246409
// MI455X (gfx1250) — matrix operand order repaired and checked by reading, not compiled, not run on hardware
//
#include <hip/hip_runtime.h>

typedef __attribute__((ext_vector_type(16))) _Float16 v16h;
typedef __attribute__((ext_vector_type(8)))  _Float16 v8h;
typedef __attribute__((ext_vector_type(8)))  float    v8f;

#define B_  32
#define S_  2048
#define D_  128
#define TQ  128           // query rows per block
#define TK  64            // key rows per iteration
#define NW  8             // waves per block
#define KHSTRIDE (D_ + 8) // f16 units: 272B row stride -> conflict-free
#define VSTRIDE  (TK + 8) // f16 units: 144B row stride -> conflict-free
#define PSTRIDE  (TK + 8) // f16 units
#define MSTRIDE  (TK + 4) // f32 units: 68 dwords -> halves 32 banks apart
#define MBUF     (TQ * MSTRIDE)

__device__ __forceinline__ v8h cvt8f(float4 a, float4 b) {
  v8h r;
  r[0]=(_Float16)a.x; r[1]=(_Float16)a.y; r[2]=(_Float16)a.z; r[3]=(_Float16)a.w;
  r[4]=(_Float16)b.x; r[5]=(_Float16)b.y; r[6]=(_Float16)b.z; r[7]=(_Float16)b.w;
  return r;
}

// async-prefetch 128B of mask row into LDS, non-temporal (one-shot stream,
// keep it out of L2 where K/V are resident)
__device__ __forceinline__ void mask_kick(const float* gm, unsigned lds) {
  unsigned long long ga = (unsigned long long)(size_t)gm;
  asm volatile(
    "global_load_async_to_lds_b128 %0, %1, off th:TH_LOAD_NT\n\t"
    "global_load_async_to_lds_b128 %0, %1, off offset:16 th:TH_LOAD_NT\n\t"
    "global_load_async_to_lds_b128 %0, %1, off offset:32 th:TH_LOAD_NT\n\t"
    "global_load_async_to_lds_b128 %0, %1, off offset:48 th:TH_LOAD_NT\n\t"
    "global_load_async_to_lds_b128 %0, %1, off offset:64 th:TH_LOAD_NT\n\t"
    "global_load_async_to_lds_b128 %0, %1, off offset:80 th:TH_LOAD_NT\n\t"
    "global_load_async_to_lds_b128 %0, %1, off offset:96 th:TH_LOAD_NT\n\t"
    "global_load_async_to_lds_b128 %0, %1, off offset:112 th:TH_LOAD_NT"
    :: "v"(lds), "v"(ga) : "memory");
}

__launch_bounds__(256, 1)
__global__ void attn_fa_kernel(const float* __restrict__ q,
                               const float* __restrict__ k,
                               const float* __restrict__ v,
                               const float* __restrict__ mask,
                               float* __restrict__ out)
{
  __shared__ _Float16 Kh[TK * KHSTRIDE];      // 17408 B, f16 K block (row-major)
  __shared__ _Float16 Vt[D_ * VSTRIDE];       // 18432 B, f16 V block (transposed [d][k])
  __shared__ _Float16 Ps[NW * 16 * PSTRIDE];  // 18432 B, per-wave P scratch
  __shared__ float    Msk[2 * MBUF];          // 69632 B, double-buffered mask tile

  const int tid  = threadIdx.x;
  const int wave = tid >> 5;
  const int lane = tid & 31;
  const int col  = lane & 15;   // N-column within 16x16 tile (and B-frag row id)
  const int half = lane >> 4;

  const int bidx = blockIdx.x;
  const int b    = bidx >> 4;            // 16 query-blocks per batch
  const int qblk = (bidx & 15) * TQ;     // block's query base
  const int q0   = qblk + wave * 16;     // wave's query base

  const float scale = 0.08838834764831845f; // 1/sqrt(128)

  // ---- persistent Q A-fragments (16 rows x 128, f32 -> f16) ----
  v16h qa[4];
  {
    const float* qrow = q + ((size_t)b * S_ + (size_t)(q0 + col)) * D_;
    #pragma unroll
    for (int c = 0; c < 4; ++c) {
      // A layout: elems 0..7 = d c*32 + half*8 .. +7 ; elems 8..15 = +16
      float4 r0a = *(const float4*)(qrow + c*32 + half*8);
      float4 r0b = *(const float4*)(qrow + c*32 + half*8 + 4);
      float4 r1a = *(const float4*)(qrow + c*32 + 16 + half*8);
      float4 r1b = *(const float4*)(qrow + c*32 + 16 + half*8 + 4);
      v8h lo = cvt8f(r0a, r0b);
      v8h hi = cvt8f(r1a, r1b);
      qa[c] = __builtin_shufflevector(lo, hi, 0,1,2,3,4,5,6,7,8,9,10,11,12,13,14,15);
    }
  }

  // ---- online-softmax state (C layout: VGPR r -> row r + 8*half) ----
  float mrow[8], lrow[8];
  v8f   Of[8];
  #pragma unroll
  for (int r = 0; r < 8; ++r) { mrow[r] = -__builtin_inff(); lrow[r] = 0.0f; }
  #pragma unroll
  for (int dt = 0; dt < 8; ++dt) {
    v8f z = {0.f,0.f,0.f,0.f,0.f,0.f,0.f,0.f};
    Of[dt] = z;
  }

  // per-thread mask-prefetch coordinates: 2 threads per query row, 32 keys each
  const int mr = tid >> 1;
  const int mc = (tid & 1) * 32;
  const float* gmrow = mask + ((size_t)b * S_ + (size_t)(qblk + mr)) * S_ + (size_t)mc;

  // ---- prologue: prefetch mask tile for kb=0 into buffer 0 ----
  mask_kick(gmrow, (unsigned)(size_t)(void*)&Msk[0 * MBUF + mr * MSTRIDE + mc]);

  #pragma unroll 1
  for (int it = 0; it < S_ / TK; ++it) {
    const int kb  = it * TK;
    const int buf = it & 1;

    __syncthreads();  // previous iteration's LDS reads finished

    // ---- K block: load f32, convert, store f16 row-major ----
    {
      const int kr = tid >> 2;
      const int c0 = (tid & 3) * 32;
      const float* gk = k + ((size_t)b * S_ + (size_t)(kb + kr)) * D_ + c0;
      #pragma unroll
      for (int j = 0; j < 4; ++j) {
        float4 x0 = *(const float4*)(gk + j*8);
        float4 x1 = *(const float4*)(gk + j*8 + 4);
        *(v8h*)&Kh[kr * KHSTRIDE + c0 + j*8] = cvt8f(x0, x1);
      }
    }

    // ---- V block: load f32, convert, store transposed f16 [d][k] ----
    {
      const int vr = tid >> 2;
      const int c0 = (tid & 3) * 32;
      const float* gv = v + ((size_t)b * S_ + (size_t)(kb + vr)) * D_ + c0;
      #pragma unroll
      for (int j = 0; j < 8; ++j) {
        float4 x = *(const float4*)(gv + j * 4);
        Vt[(c0 + j*4 + 0) * VSTRIDE + vr] = (_Float16)x.x;
        Vt[(c0 + j*4 + 1) * VSTRIDE + vr] = (_Float16)x.y;
        Vt[(c0 + j*4 + 2) * VSTRIDE + vr] = (_Float16)x.z;
        Vt[(c0 + j*4 + 3) * VSTRIDE + vr] = (_Float16)x.w;
      }
    }

    // ---- prefetch next iteration's mask tile (wraps on last iter: keeps the
    //      asynccnt discipline uniform; 16MB redundant vs 537MB total) ----
    {
      const int kbn  = (kb + TK) & (S_ - 1);
      const int nbuf = buf ^ 1;
      mask_kick(gmrow + kbn,
                (unsigned)(size_t)(void*)&Msk[nbuf * MBUF + mr * MSTRIDE + mc]);
    }

    // async loads complete in order: <=8 outstanding => current buffer done
    asm volatile("s_wait_asynccnt 8" ::: "memory");
    __syncthreads();  // K/V stores + every wave's mask prefetch visible

    // ---- S = Q @ K^T : 4 column tiles of 16 keys ----
    float p[4][8];
    float tmax[8];
    #pragma unroll
    for (int r = 0; r < 8; ++r) tmax[r] = -__builtin_inff();

    const float* mbuf = &Msk[buf * MBUF + (wave * 16) * MSTRIDE];

    #pragma unroll
    for (int t = 0; t < 4; ++t) {
      v8f acc = {0.f,0.f,0.f,0.f,0.f,0.f,0.f,0.f};
      #pragma unroll
      for (int c = 0; c < 4; ++c) {
        // B-frag: lane holds key row 'col', elems 0..7 = d c*32 + half*8 .. +7 ; elems 8..15 = +16
        const _Float16* kp = &Kh[(t*16 + col) * KHSTRIDE + c*32 + half*8];
        v8h b0 = *(const v8h*)(kp);
        v8h b1 = *(const v8h*)(kp + 16);
        v16h bf = __builtin_shufflevector(b0, b1, 0,1,2,3,4,5,6,7,8,9,10,11,12,13,14,15);
        acc = __builtin_amdgcn_wmma_f32_16x16x32_f16(
            false, qa[c], false, bf, (short)0, acc, false, false);
      }
      // scale + multiplicative mask from the prefetched LDS tile
      #pragma unroll
      for (int r = 0; r < 8; ++r) {
        float mm = mbuf[(r + 8*half) * MSTRIDE + t*16 + col];
        float s  = acc[r] * scale * mm;
        p[t][r]  = s;
        tmax[r]  = fmaxf(tmax[r], s);
      }
    }

    // ---- online softmax update (reduce across the 16-lane half-group) ----
    #pragma unroll
    for (int r = 0; r < 8; ++r) {
      float m = tmax[r];
      m = fmaxf(m, __shfl_xor(m, 1));
      m = fmaxf(m, __shfl_xor(m, 2));
      m = fmaxf(m, __shfl_xor(m, 4));
      m = fmaxf(m, __shfl_xor(m, 8));
      float mnew  = fmaxf(mrow[r], m);
      float alpha = __expf(mrow[r] - mnew);
      mrow[r] = mnew;
      float s = 0.f;
      #pragma unroll
      for (int t = 0; t < 4; ++t) {
        float e = __expf(p[t][r] - mnew);
        p[t][r] = e;
        s += e;
      }
      s += __shfl_xor(s, 1);
      s += __shfl_xor(s, 2);
      s += __shfl_xor(s, 4);
      s += __shfl_xor(s, 8);
      lrow[r] = lrow[r] * alpha + s;
      #pragma unroll
      for (int dt = 0; dt < 8; ++dt) Of[dt][r] *= alpha;
    }

    // ---- P: C-layout -> A-layout via per-wave LDS scratch ----
    _Float16* pw = &Ps[wave * 16 * PSTRIDE];
    #pragma unroll
    for (int t = 0; t < 4; ++t)
      #pragma unroll
      for (int r = 0; r < 8; ++r)
        pw[(r + 8*half) * PSTRIDE + t*16 + col] = (_Float16)p[t][r];
    asm volatile("s_wait_dscnt 0" ::: "memory");

    v16h pa[2];
    #pragma unroll
    for (int c = 0; c < 2; ++c) {
      const _Float16* pr = &pw[col * PSTRIDE + c*32];
      v8h lo = *(const v8h*)(pr + half*8);
      v8h hi = *(const v8h*)(pr + 16 + half*8);
      pa[c] = __builtin_shufflevector(lo, hi, 0,1,2,3,4,5,6,7,8,9,10,11,12,13,14,15);
    }

    // ---- O += P @ V : 8 d-tiles x 2 contraction chunks ----
    #pragma unroll
    for (int dt = 0; dt < 8; ++dt) {
      v8f acc = Of[dt];
      #pragma unroll
      for (int c = 0; c < 2; ++c) {
        const _Float16* vp = &Vt[(dt*16 + col) * VSTRIDE + c*32 + half*8];
        v8h b0 = *(const v8h*)(vp);
        v8h b1 = *(const v8h*)(vp + 16);
        v16h bf = __builtin_shufflevector(b0, b1, 0,1,2,3,4,5,6,7,8,9,10,11,12,13,14,15);
        acc = __builtin_amdgcn_wmma_f32_16x16x32_f16(
            false, pa[c], false, bf, (short)0, acc, false, false);
      }
      Of[dt] = acc;
    }
  }

  // drain our final (wrapped) prefetch before wave exit
  asm volatile("s_wait_asynccnt 0" ::: "memory");

  // ---- epilogue: normalize and store ----
  #pragma unroll
  for (int r = 0; r < 8; ++r) {
    float inv = 1.0f / lrow[r];
    float* orow = out + ((size_t)b * S_ + (size_t)(q0 + r + 8*half)) * D_;
    #pragma unroll
    for (int dt = 0; dt < 8; ++dt)
      orow[dt*16 + col] = Of[dt][r] * inv;
  }
}

extern "C" void kernel_launch(void* const* d_in, const int* in_sizes, int n_in,
                              void* d_out, int out_size, void* d_ws, size_t ws_size,
                              hipStream_t stream) {
  (void)in_sizes; (void)n_in; (void)out_size; (void)d_ws; (void)ws_size;
  const float* q    = (const float*)d_in[0];
  const float* k    = (const float*)d_in[1];
  const float* v    = (const float*)d_in[2];
  const float* mask = (const float*)d_in[3];
  float* out = (float*)d_out;

  dim3 grid(B_ * (S_ / TQ));   // 32 * 16 = 512 blocks
  dim3 block(256);             // 8 waves (wave32)
  attn_fa_kernel<<<grid, block, 0, stream>>>(q, k, v, mask, out);
}
